// q_global_59760174957039
// MI455X (gfx1250) — matrix operand order repaired and checked by reading, not compiled, not run on hardware
//
#include <hip/hip_runtime.h>
#include <cstdint>
#include <cstddef>

// ---------------------------------------------------------------------------
// QMIX mixer + 4 fused hypernets, bf16 WMMA (gfx1250 / CDNA5).
// One workgroup = 32 samples (2 WMMA row-tiles). B fragments and A fragments
// are software-pipelined with parity-indexed double buffers (no register
// copies, partial s_wait_loadcnt / s_wait_dscnt). Dynamic LDS = 108.5 KB.
// ---------------------------------------------------------------------------

typedef __attribute__((ext_vector_type(16))) __bf16 v16bf;
typedef __attribute__((ext_vector_type(8)))  float  v8f;

#define T_  32
#define B_  2048
#define N_  3
#define D_  512
#define HH_ 256
#define H_  64
#define S_  (T_ * B_)     // 65536 samples
#define M_  32            // samples per workgroup
#define NT1 64            // hidden column tiles (1024 / 16)
#define KT1 16            // layer-1 K tiles   (512 / 32)
#define NT2 21            // layer-2 out tiles (336 / 16), cols 321..335 pad
#define KT2 8             // layer-2 K tiles   (256 / 32)
#define OW  336           // padded layer-2 output width

// LDS map (dynamic): [0, 43008) = ldsA (32x512 bf16, 32KB) aliased with
// ldsO (32x336 f32, 42KB); [43008, 108544) = ldsH (32x1024 bf16, 64KB)
#define LDS_H_OFF  43008
#define LDS_BYTES  (43008 + 65536)

__device__ __forceinline__ unsigned short f2bf(float f) {
  unsigned u = __float_as_uint(f);
  u += 0x7FFFu + ((u >> 16) & 1u);   // round-to-nearest-even
  return (unsigned short)(u >> 16);
}

union Frag { uint4 q[2]; v16bf v; };

// ---------------- weight repack: layer 1 (4x 512x256 fp32 -> fragments) -----
struct PackW1Args { const float* W[4]; unsigned short* dst; };

__global__ __launch_bounds__(32) void pack_w1(PackW1Args a) {
  int f    = blockIdx.x;            // fragment = nt*KT1 + kt, 1024 total
  int nt   = f >> 4, kt = f & 15;
  int lane = threadIdx.x;
  int c    = nt * 16 + (lane & 15); // global hidden column 0..1023
  const float* W = a.W[c >> 8];     // which hypernet's W1
  int col  = c & 255;
  int kb   = kt * 32 + ((lane >> 4) << 3); // B layout: lanes16-31 take K+8 (table order)
  unsigned v[8];
#pragma unroll
  for (int j = 0; j < 8; ++j) {
    unsigned lo = f2bf(W[(kb + 2 * (j & 3) + ((j >> 2) << 4)    ) * HH_ + col]);
    unsigned hi = f2bf(W[(kb + 2 * (j & 3) + ((j >> 2) << 4) + 1) * HH_ + col]);
    v[j] = lo | (hi << 16);
  }
  uint4* d = (uint4*)(a.dst + (size_t)f * 512 + lane * 16);
  d[0] = make_uint4(v[0], v[1], v[2], v[3]);
  d[1] = make_uint4(v[4], v[5], v[6], v[7]);
}

// ---------------- weight repack: layer 2 (block-diagonal) -------------------
struct PackW2Args { const float* W[4]; unsigned short* dst; };

__global__ __launch_bounds__(32) void pack_w2(PackW2Args a) {
  int f    = blockIdx.x;            // fragment = t*KT2 + kt, 168 total
  int t    = f >> 3, kt = f & 7;
  int lane = threadIdx.x;
  int h, width, cbase;
  if      (t < 12) { h = 0; width = N_ * H_; cbase = 0;   }  // w1: 192 cols
  else if (t < 16) { h = 1; width = H_;      cbase = 192; }  // b1: 64 cols
  else if (t < 20) { h = 2; width = H_;      cbase = 256; }  // w2: 64 cols
  else             { h = 3; width = 1;       cbase = 320; }  // b2: 1 col
  const float* W = a.W[h];
  int lc = t * 16 + (lane & 15) - cbase;    // local output column
  int kb = kt * 32 + ((lane >> 4) << 3);
  unsigned v[8];
#pragma unroll
  for (int j = 0; j < 8; ++j) {
    float f0 = (lc < width) ? W[(kb + 2 * (j & 3) + ((j >> 2) << 4)    ) * width + lc] : 0.0f;
    float f1 = (lc < width) ? W[(kb + 2 * (j & 3) + ((j >> 2) << 4) + 1) * width + lc] : 0.0f;
    v[j] = (unsigned)f2bf(f0) | ((unsigned)f2bf(f1) << 16);
  }
  uint4* d = (uint4*)(a.dst + (size_t)f * 512 + lane * 16);
  d[0] = make_uint4(v[0], v[1], v[2], v[3]);
  d[1] = make_uint4(v[4], v[5], v[6], v[7]);
}

// ---------------- fused hypernet + mixer kernel -----------------------------
struct MixArgs {
  const float* q;                 // (T,B,N)
  const float* states;            // (T,B,D)
  const float* b1[4];             // layer-1 biases (256 each)
  const float* b2[4];             // layer-2 biases (192,64,64,1)
  const unsigned short* packed1;  // 1024 fragments x 1KB
  const unsigned short* packed2;  // 168 fragments x 1KB
  float* out;                     // (T,B,N)
};

__global__ __launch_bounds__(256) void qmix_fused(MixArgs A) {
  extern __shared__ __align__(16) unsigned char smem[];
  unsigned short* ldsA = (unsigned short*)smem;               // phase 0/1
  float*          ldsO = (float*)smem;                        // phase 2/3
  unsigned short* ldsH = (unsigned short*)(smem + LDS_H_OFF); // hidden bf16

  const int tid  = threadIdx.x;
  const int wave = tid >> 5;
  const int lane = tid & 31;
  const int row  = lane & 15;     // WMMA A row / C column within tile
  const int sel  = lane >> 4;     // lane-half selector for A/C layouts
  const size_t sbase = (size_t)blockIdx.x * M_;

  // ---- phase 0: stage 32x512 fp32 states tile into LDS as bf16 ----
  const float4* st4 = (const float4*)(A.states + sbase * D_);
#pragma unroll
  for (int i = 0; i < 16; ++i) {
    int idx  = tid + i * 256;     // float4 index 0..4095 (contiguous tile)
    float4 v = st4[idx];
    unsigned short* d = ldsA + idx * 4;
    ((uint2*)d)[0] = make_uint2(
        (unsigned)f2bf(v.x) | ((unsigned)f2bf(v.y) << 16),
        (unsigned)f2bf(v.z) | ((unsigned)f2bf(v.w) << 16));
  }
  __syncthreads();

  // ---- phase 1: hidden(32x1024) = relu(states @ W1cat + b1cat) ----
  // wave w owns hidden column tiles w*8 .. w*8+7; 2 sample sub-tiles.
  v8f acc0[8], acc1[8];
#pragma unroll
  for (int n = 0; n < 8; ++n) { acc0[n] = (v8f)0.0f; acc1[n] = (v8f)0.0f; }

  const unsigned short* pbase = A.packed1 + (size_t)(wave * 8) * KT1 * 512;
  // fragment (kt, n) lives at pbase + (n*KT1 + kt)*512 halfwords
  const unsigned short* abase = ldsA + row * 512 + sel * 8;

  Frag bbuf[2];                   // parity-indexed B double buffer (n & 1)
  Frag a0[2], a1[2];              // parity-indexed A double buffer (kt & 1)
  {
    const uint4* bp = (const uint4*)(pbase + (size_t)lane * 16);
    bbuf[0].q[0] = bp[0]; bbuf[0].q[1] = bp[1];
    a0[0].q[0] = *(const uint4*)abase;
    a0[0].q[1] = *(const uint4*)(abase + 16);
    a1[0].q[0] = *(const uint4*)(abase + 16 * 512);
    a1[0].q[1] = *(const uint4*)(abase + 16 * 512 + 16);
  }
  for (int kt = 0; kt < KT1; ++kt) {
    const int p = kt & 1;
    if (kt + 1 < KT1) {           // prefetch next A fragments (LDS) early
      const unsigned short* ap = abase + (kt + 1) * 32;
      a0[p ^ 1].q[0] = *(const uint4*)ap;
      a0[p ^ 1].q[1] = *(const uint4*)(ap + 16);
      a1[p ^ 1].q[0] = *(const uint4*)(ap + 16 * 512);
      a1[p ^ 1].q[1] = *(const uint4*)(ap + 16 * 512 + 16);
    }
#pragma unroll
    for (int n = 0; n < 8; ++n) {
      const int cur = n & 1;      // consumed buffer
      const int nxt = cur ^ 1;    // refill buffer (no copies: parity rotation)
      int nn  = (n == 7) ? 0 : n + 1;
      int nkt = (n == 7) ? kt + 1 : kt;
      if (nkt < KT1) {
        const uint4* bp = (const uint4*)(pbase + (size_t)(nn * KT1 + nkt) * 512 +
                                         (size_t)lane * 16);
        bbuf[nxt].q[0] = bp[0]; bbuf[nxt].q[1] = bp[1];
      }
      acc0[n] = __builtin_amdgcn_wmma_f32_16x16x32_bf16(
          false, a0[p].v, false, bbuf[cur].v, (short)0, acc0[n], false, false);
      acc1[n] = __builtin_amdgcn_wmma_f32_16x16x32_bf16(
          false, a1[p].v, false, bbuf[cur].v, (short)0, acc1[n], false, false);
    }
  }

  // epilogue: +bias, relu, bf16 -> ldsH (row-major 32x1024)
#pragma unroll
  for (int n = 0; n < 8; ++n) {
    int c = (wave * 8 + n) * 16 + row;      // global hidden column
    float bias = A.b1[c >> 8][c & 255];     // per column; same for all 8 rows r
#pragma unroll
    for (int r = 0; r < 8; ++r) {
      int m = r + (sel << 3);               // C layout: M = r + 8*sel
      float v0 = fmaxf(acc0[n][r] + bias, 0.0f);
      float v1 = fmaxf(acc1[n][r] + bias, 0.0f);
      ldsH[m * 1024 + c]        = f2bf(v0);
      ldsH[(m + 16) * 1024 + c] = f2bf(v1);
    }
  }
  __syncthreads();

  // ---- phase 2: block-diagonal layer 2 -> ldsO (32x336 f32) ----
  for (int t = wave; t < NT2; t += 8) {
    int h  = (t < 12) ? 0 : (t < 16) ? 1 : (t < 20) ? 2 : 3;
    int kb = h << 8;                        // hidden column base for block
    const unsigned short* hbase = ldsH + row * 1024 + kb + sel * 8;
    const unsigned short* p2base = A.packed2 + (size_t)(t * 8) * 512;

    v8f c20 = (v8f)0.0f, c21 = (v8f)0.0f;
    Frag b2f[2];
    {
      const uint4* bp = (const uint4*)(p2base + (size_t)lane * 16);
      b2f[0].q[0] = bp[0]; b2f[0].q[1] = bp[1];
    }
#pragma unroll
    for (int kt = 0; kt < KT2; ++kt) {
      const int cur = kt & 1, nxt = cur ^ 1;
      if (kt + 1 < KT2) {
        const uint4* bp = (const uint4*)(p2base + (size_t)(kt + 1) * 512 +
                                         (size_t)lane * 16);
        b2f[nxt].q[0] = bp[0]; b2f[nxt].q[1] = bp[1];
      }
      Frag af0, af1;
      const unsigned short* ap = hbase + kt * 32;
      af0.q[0] = *(const uint4*)ap;
      af0.q[1] = *(const uint4*)(ap + 16);
      af1.q[0] = *(const uint4*)(ap + 16 * 1024);
      af1.q[1] = *(const uint4*)(ap + 16 * 1024 + 16);
      c20 = __builtin_amdgcn_wmma_f32_16x16x32_bf16(
          false, af0.v, false, b2f[cur].v, (short)0, c20, false, false);
      c21 = __builtin_amdgcn_wmma_f32_16x16x32_bf16(
          false, af1.v, false, b2f[cur].v, (short)0, c21, false, false);
    }
    int col = t * 16 + row;
    float bias = 0.0f;
    if      (col < 192)  bias = A.b2[0][col];
    else if (col < 256)  bias = A.b2[1][col - 192];
    else if (col < 320)  bias = A.b2[2][col - 256];
    else if (col == 320) bias = A.b2[3][0];
#pragma unroll
    for (int r = 0; r < 8; ++r) {
      int m = r + (sel << 3);
      ldsO[m * OW + col]        = c20[r] + bias;
      ldsO[(m + 16) * OW + col] = c21[r] + bias;
    }
  }
  __syncthreads();

  // ---- phase 3: per-sample mixer, 96 lanes = 32 samples x 3 rotations ----
  if (tid < M_ * N_) {
    int s = tid / 3, rot = tid % 3;
    const float* rowp = ldsO + s * OW;
    const float* qp   = A.q + (sbase + s) * N_;
    float q0 = qp[0], q1 = qp[1], q2 = qp[2];
    float qa, qb, qc;                       // PERM rows {0,1,2},{1,0,2},{2,0,1}
    if      (rot == 0) { qa = q0; qb = q1; qc = q2; }
    else if (rot == 1) { qa = q1; qb = q0; qc = q2; }
    else               { qa = q2; qb = q0; qc = q1; }
    float accm = 0.0f;
#pragma unroll 4
    for (int hh = 0; hh < H_; ++hh) {
      float pre = qa * fabsf(rowp[hh]) + qb * fabsf(rowp[64 + hh]) +
                  qc * fabsf(rowp[128 + hh]) + rowp[192 + hh];
      float e = (pre > 0.0f) ? pre : (__expf(pre) - 1.0f);  // elu
      accm += e * fabsf(rowp[256 + hh]);
    }
    accm += rowp[320];
    A.out[(sbase + s) * N_ + rot] = accm;
  }
}

// ---------------------------------------------------------------------------
extern "C" void kernel_launch(void* const* d_in, const int* in_sizes, int n_in,
                              void* d_out, int out_size, void* d_ws, size_t ws_size,
                              hipStream_t stream) {
  (void)in_sizes; (void)n_in; (void)out_size; (void)ws_size;

  unsigned short* packed1 = (unsigned short*)d_ws;                       // 1 MiB
  unsigned short* packed2 =
      (unsigned short*)((char*)d_ws + (size_t)NT1 * KT1 * 1024);         // 168 KiB

  PackW1Args p1;
  p1.W[0] = (const float*)d_in[2];   // hw1_W1
  p1.W[1] = (const float*)d_in[6];   // hb1_W1
  p1.W[2] = (const float*)d_in[10];  // hw2_W1
  p1.W[3] = (const float*)d_in[14];  // hb2_W1
  p1.dst  = packed1;
  pack_w1<<<NT1 * KT1, 32, 0, stream>>>(p1);

  PackW2Args p2;
  p2.W[0] = (const float*)d_in[4];   // hw1_W2 (256x192)
  p2.W[1] = (const float*)d_in[8];   // hb1_W2 (256x64)
  p2.W[2] = (const float*)d_in[12];  // hw2_W2 (256x64)
  p2.W[3] = (const float*)d_in[16];  // hb2_W2 (256x1)
  p2.dst  = packed2;
  pack_w2<<<NT2 * KT2, 32, 0, stream>>>(p2);

  MixArgs m;
  m.q      = (const float*)d_in[0];
  m.states = (const float*)d_in[1];
  m.b1[0] = (const float*)d_in[3];  m.b1[1] = (const float*)d_in[7];
  m.b1[2] = (const float*)d_in[11]; m.b1[3] = (const float*)d_in[15];
  m.b2[0] = (const float*)d_in[5];  m.b2[1] = (const float*)d_in[9];
  m.b2[2] = (const float*)d_in[13]; m.b2[3] = (const float*)d_in[17];
  m.packed1 = packed1;
  m.packed2 = packed2;
  m.out     = (float*)d_out;
  qmix_fused<<<S_ / M_, 256, LDS_BYTES, stream>>>(m);
}
